// TCOVISCriterion_19456201851405
// MI455X (gfx1250) — matrix operand order repaired and checked by reading, not compiled, not run on hardware
//
#include <hip/hip_runtime.h>
#include <hip/hip_bf16.h>

typedef __attribute__((ext_vector_type(16))) _Float16 v16h;
typedef __attribute__((ext_vector_type(8)))  _Float16 v8h;
typedef __attribute__((ext_vector_type(8)))  float    v8f;

// Problem constants (fixed by reference setup_inputs)
constexpr int B = 2, T = 6, N = 100, H = 128, W = 128, M = 20, D = 256;
constexpr int HW  = H * W;          // 16384
constexpr int TBM = T * B * M;      // 240
constexpr int MP  = 32;             // M padded to 2x16 for WMMA tiles
constexpr int SPLIT = 8;            // blocks per (t,b,m) mask slice
constexpr int CHUNK = HW / SPLIT;   // 2048 elements per block

constexpr float TEMPERATURE = 0.1f;
constexpr float ALPHA = 5.0f;
constexpr float Wm = 5.0f, Wd = 5.0f, Wmt = 1.0f, Wc = 0.5f;

// Workspace layout:
//   float acc[8]            : 0=bce_sum 1=dice_sum 2=match_sum 3=ctr_sum
//   float tbm_part[240*3]   : per-(t,b,m) {sum_sig, sum_sig*tgt, sum_tgt}
//   _Float16 g[T*B*MP*D]    : normalized gathered embeddings (rows 20..31 zero), at +4096B
constexpr int NF_ZERO = 8 + TBM * 3;                 // floats to zero
constexpr int NG_HALF = T * B * MP * D;              // 98304 halves
constexpr int NG_WORD = NG_HALF / 2;                 // 49152 u32 words

// ---------------------------------------------------------------- reductions
__device__ inline float block_sum256(float v, float* red8) {
#pragma unroll
  for (int o = 16; o > 0; o >>= 1) v += __shfl_down(v, o, 32);
  const int lane = threadIdx.x & 31, wid = threadIdx.x >> 5;
  __syncthreads();                       // protect red8 reuse
  if (lane == 0) red8[wid] = v;
  __syncthreads();
  float r = 0.f;
  if (wid == 0) {
    r = (lane < 8) ? red8[lane] : 0.f;
#pragma unroll
    for (int o = 4; o > 0; o >>= 1) r += __shfl_down(r, o, 32);
  }
  return r;                              // valid in thread 0
}

// ---------------------------------------------------------------- init
__global__ __launch_bounds__(256) void k_init(float* ws_f, unsigned* g_u) {
  const int i = blockIdx.x * blockDim.x + threadIdx.x;
  if (i < NF_ZERO) ws_f[i] = 0.f;
  if (i < NG_WORD) g_u[i] = 0u;
}

// ---------------------------------------------------------------- mask partials
__device__ inline void bce_elem(float s, float tg, float& abce, float& asig,
                                float& ast, float& atg) {
  const float e   = __expf(-fabsf(s));
  const float bce = fmaxf(s, 0.f) - s * tg + __logf(1.f + e);   // log1p(exp(-|s|))
  const float sg  = 1.f / (1.f + __expf(-s));
  abce += bce; asig += sg; ast += sg * tg; atg += tg;
}

__global__ __launch_bounds__(256) void k_mask_partial(
    const float* __restrict__ pred_masks, const float* __restrict__ gt_masks,
    const int* __restrict__ perm, float* __restrict__ acc,
    float* __restrict__ tbm_part) {
  const int blk   = blockIdx.x;
  const int piece = blk % SPLIT;
  const int tbm   = blk / SPLIT;          // (t*B + b)*M + m
  const int m  = tbm % M;
  const int tb = tbm / M;
  const int b  = tb % B;
  const int t  = tb / B;
  const int n  = perm[tbm];               // match_perm is (T,B,M) contiguous

  const float4* __restrict__ src4 =
      reinterpret_cast<const float4*>(pred_masks + ((size_t)((b * T + t) * N) + n) * HW)
      + (size_t)piece * (CHUNK / 4);
  const float4* __restrict__ tgt4 =
      reinterpret_cast<const float4*>(gt_masks + ((size_t)((b * T + t) * M) + m) * HW)
      + (size_t)piece * (CHUNK / 4);

  float abce = 0.f, asig = 0.f, ast = 0.f, atg = 0.f;
  for (int i = threadIdx.x; i < CHUNK / 4; i += 256) {
    const float4 s = src4[i];
    const float4 g = tgt4[i];
    bce_elem(s.x, g.x, abce, asig, ast, atg);
    bce_elem(s.y, g.y, abce, asig, ast, atg);
    bce_elem(s.z, g.z, abce, asig, ast, atg);
    bce_elem(s.w, g.w, abce, asig, ast, atg);
  }

  __shared__ float red8[8];
  const float rb = block_sum256(abce, red8);
  const float rs = block_sum256(asig, red8);
  const float rt = block_sum256(ast,  red8);
  const float rg = block_sum256(atg,  red8);
  if (threadIdx.x == 0) {
    atomicAdd(&acc[0], rb);
    atomicAdd(&tbm_part[tbm * 3 + 0], rs);
    atomicAdd(&tbm_part[tbm * 3 + 1], rt);
    atomicAdd(&tbm_part[tbm * 3 + 2], rg);
  }
}

// ---------------------------------------------------------------- dice finalize
__global__ __launch_bounds__(256) void k_dice(const float* __restrict__ tbm_part,
                                              float* __restrict__ acc) {
  const int i = blockIdx.x * blockDim.x + threadIdx.x;
  if (i < TBM) {
    const float sig = tbm_part[i * 3 + 0];
    const float st  = tbm_part[i * 3 + 1];
    const float tg  = tbm_part[i * 3 + 2];
    const float dice = 1.f - 2.f * st / (sig + tg + 1e-6f);
    atomicAdd(&acc[1], dice);
  }
}

// ---------------------------------------------------------------- embeddings
__global__ __launch_bounds__(256) void k_emb(const float* __restrict__ pred_embs,
                                             const int* __restrict__ perm,
                                             _Float16* __restrict__ g) {
  const int tbm = blockIdx.x;             // (t*B + b)*M + m
  const int m  = tbm % M;
  const int tb = tbm / M;
  const int b  = tb % B;
  const int t  = tb / B;
  const int n  = perm[tbm];

  const float e = pred_embs[((size_t)((b * T + t) * N) + n) * D + threadIdx.x];
  __shared__ float red8[8];
  const float ss = block_sum256(e * e, red8);
  __shared__ float inv_s;
  if (threadIdx.x == 0) inv_s = 1.f / fmaxf(sqrtf(ss), 1e-12f);
  __syncthreads();
  g[((size_t)((t * B + b) * MP) + m) * D + threadIdx.x] = (_Float16)(e * inv_s);
}

// ---------------------------------------------------------------- WMMA sim + losses
__device__ inline v16h load_a_tile(const _Float16* __restrict__ base, int rowBase,
                                   int k, int l15, int half) {
  // A 16x32 f16 layout (ISA 7.12.2): lanes 0-15 hold K 0..7 & 16..23,
  // lanes 16-31 hold K 8..15 & 24..31, row M = lane&15.
  const _Float16* p = base + (size_t)(rowBase + l15) * D + k + half * 8;
  const v8h lo = *(const v8h*)(p);
  const v8h hi = *(const v8h*)(p + 16);
  v16h a;
#pragma unroll
  for (int i = 0; i < 8; ++i) { a[i] = lo[i]; a[8 + i] = hi[i]; }
  return a;
}

__device__ inline v16h load_b_tile(const _Float16* __restrict__ base, int colBase,
                                   int k, int l15, int half) {
  // B 32x16 f16 layout: lanes 0-15 hold K 0..7 & 16..23, lanes 16-31 hold K 8..15 & 24..31,
  // column N = lane&15.  B[k][n] = nxt_emb[n][k] -> two 16B chunks per lane.
  const _Float16* p = base + (size_t)(colBase + l15) * D + k + half * 8;
  const v8h lo = *(const v8h*)(p);
  const v8h hi = *(const v8h*)(p + 16);
  v16h bv;
#pragma unroll
  for (int i = 0; i < 8; ++i) { bv[i] = lo[i]; bv[8 + i] = hi[i]; }
  return bv;
}

__global__ __launch_bounds__(32) void k_sim(const _Float16* __restrict__ g,
                                            float* __restrict__ acc) {
  const int tb = blockIdx.x;              // (t, b), t in 0..T-2
  const int b  = tb % B;
  const int t  = tb / B;
  const _Float16* curr = g + (size_t)((t * B) + b) * MP * D;
  const _Float16* nxt  = g + (size_t)(((t + 1) * B) + b) * MP * D;

  const int lane = threadIdx.x;
  const int l15  = lane & 15;
  const int half = lane >> 4;

  v8f c00 = {}, c01 = {}, c10 = {}, c11 = {};
#pragma unroll
  for (int k = 0; k < D; k += 32) {
    const v16h a0 = load_a_tile(curr, 0,  k, l15, half);
    const v16h a1 = load_a_tile(curr, 16, k, l15, half);
    const v16h b0 = load_b_tile(nxt, 0,  k, l15, half);
    const v16h b1 = load_b_tile(nxt, 16, k, l15, half);
    c00 = __builtin_amdgcn_wmma_f32_16x16x32_f16(false, a0, false, b0, (short)0, c00, false, false);
    c01 = __builtin_amdgcn_wmma_f32_16x16x32_f16(false, a0, false, b1, (short)0, c01, false, false);
    c10 = __builtin_amdgcn_wmma_f32_16x16x32_f16(false, a1, false, b0, (short)0, c10, false, false);
    c11 = __builtin_amdgcn_wmma_f32_16x16x32_f16(false, a1, false, b1, (short)0, c11, false, false);
  }

  // C/D layout: VGPR v -> M = half*8 + v, N = lane&15.
  __shared__ float sim[MP][MP];
#pragma unroll
  for (int v = 0; v < 8; ++v) {
    const int r = half * 8 + v;
    sim[r][l15]           = c00[v];
    sim[r][16 + l15]      = c01[v];
    sim[16 + r][l15]      = c10[v];
    sim[16 + r][16 + l15] = c11[v];
  }
  __syncthreads();

  float lm = 0.f, lc = 0.f;
  if (lane < M) {
    const float diag = sim[lane][lane];
    float mx = -1e30f;
#pragma unroll
    for (int j = 0; j < M; ++j) mx = fmaxf(mx, sim[lane][j]);
    float sa = 0.f, sc = 0.f;
#pragma unroll
    for (int j = 0; j < M; ++j) {
      const float x = sim[lane][j] - mx;
      sa += __expf(ALPHA * x);
      sc += __expf(x / TEMPERATURE);
    }
    lm = -(ALPHA * (diag - mx) - __logf(sa));
    lc = -((diag - mx) / TEMPERATURE - __logf(sc));
  }
#pragma unroll
  for (int o = 16; o > 0; o >>= 1) {
    lm += __shfl_down(lm, o, 32);
    lc += __shfl_down(lc, o, 32);
  }
  if (lane == 0) {
    atomicAdd(&acc[2], lm);
    atomicAdd(&acc[3], lc);
  }
}

// ---------------------------------------------------------------- finalize
__global__ void k_final(const float* __restrict__ acc, float* __restrict__ out) {
  if (threadIdx.x == 0 && blockIdx.x == 0) {
    const float loss_mask  = acc[0] / (float)((size_t)TBM * HW);
    const float loss_dice  = acc[1] / (float)TBM;
    const float loss_match = acc[2] / (float)((T - 1) * B * M);
    const float loss_ctr   = acc[3] / (float)((T - 1) * B * M);
    const float total = Wm * loss_mask + Wd * loss_dice + Wmt * loss_match + Wc * loss_ctr;
    out[0] = loss_mask;
    out[1] = loss_dice;
    out[2] = loss_ctr;
    out[3] = loss_match;
    out[4] = total;
  }
}

// ---------------------------------------------------------------- launch
extern "C" void kernel_launch(void* const* d_in, const int* in_sizes, int n_in,
                              void* d_out, int out_size, void* d_ws, size_t ws_size,
                              hipStream_t stream) {
  const float* pred_masks = (const float*)d_in[0];
  const float* pred_embs  = (const float*)d_in[1];
  const float* gt_masks   = (const float*)d_in[2];
  const int*   perm       = (const int*)d_in[3];

  float*    acc      = (float*)d_ws;            // [0..7]
  float*    tbm_part = acc + 8;                  // 240*3 floats
  _Float16* g        = (_Float16*)((char*)d_ws + 4096);
  unsigned* g_u      = (unsigned*)g;

  k_init<<<(NG_WORD + 255) / 256, 256, 0, stream>>>(acc, g_u);
  k_mask_partial<<<TBM * SPLIT, 256, 0, stream>>>(pred_masks, gt_masks, perm, acc, tbm_part);
  k_dice<<<1, 256, 0, stream>>>(tbm_part, acc);
  k_emb<<<TBM, 256, 0, stream>>>(pred_embs, perm, g);
  k_sim<<<(T - 1) * B, 32, 0, stream>>>(g, acc);
  k_final<<<1, 1, 0, stream>>>(acc, (float*)d_out);
}
